// StyleGAN_44487271252266
// MI455X (gfx1250) — matrix operand order repaired and checked by reading, compiled, not run on hardware
//
#include <hip/hip_runtime.h>
#include <hip/hip_bf16.h>

typedef __attribute__((ext_vector_type(16))) __bf16 v16bf;
typedef __attribute__((ext_vector_type(8)))  float  v8f;

#define LEAK 0.2f
#define EPSV 1e-8f

static __device__ __forceinline__ unsigned sg_pack_bf16x2(float a, float b) {
  union { __bf16 h[2]; unsigned u; } p;
  p.h[0] = (__bf16)a;
  p.h[1] = (__bf16)b;
  return p.u;        // lowers to v_cvt_pk_bf16_f32
}

// ---------------------------------------------------------------------------
// Kernel 1: styles + demod scales.
// scale[(i*2+j)*4 + n][o] = s * rsqrt(s*s*rownorm2 + eps),
//   s = style[n] . aff_w[i][o] + aff_b[i][o], rownorm2 = sum_c mod_w[i][o][c]^2
// ---------------------------------------------------------------------------
__global__ __launch_bounds__(256) void sg_styles_scales(
    const float* __restrict__ style,
    const float* __restrict__ aff1_w, const float* __restrict__ aff1_b,
    const float* __restrict__ mod1_w,
    const float* __restrict__ aff2_w, const float* __restrict__ aff2_b,
    const float* __restrict__ mod2_w,
    float* __restrict__ scales) {
  int idx = blockIdx.x * 256 + threadIdx.x;       // 5*2*512 = 5120 threads
  if (idx >= 5 * 2 * 512) return;
  int o = idx & 511;
  int j = (idx >> 9) & 1;
  int i = idx >> 10;
  const float* aw = (j ? aff2_w : aff1_w) + ((size_t)(i * 512 + o)) * 512;
  const float* mw = (j ? mod2_w : mod1_w) + ((size_t)(i * 512 + o)) * 512;
  float ab = (j ? aff2_b : aff1_b)[i * 512 + o];
  float rn = 0.f;
  for (int c = 0; c < 512; ++c) rn += mw[c] * mw[c];
  for (int n = 0; n < 4; ++n) {
    float s = ab;
    const float* st = style + n * 512;
    for (int c = 0; c < 512; ++c) s += st[c] * aw[c];
    scales[((size_t)((i * 2 + j) * 4 + n)) * 512 + o] = s * rsqrtf(s * s * rn + EPSV);
  }
}

// ---------------------------------------------------------------------------
// Kernel 2: convert mod weights fp32 -> bf16 in WMMA A-matrix tile layout.
// Tile block (mtile = o/16, kt = k/32): [lane][j], lane = (o&15)+16*((kk>>3)&1),
// j = (kk&7)+8*(kk>=16)  (matches 16-bit A 16x32 layout, ISA 7.12.2)
// ---------------------------------------------------------------------------
__global__ __launch_bounds__(256) void sg_convert_weights(
    const float* __restrict__ mod1_w, const float* __restrict__ mod2_w,
    __bf16* __restrict__ wb) {
  int idx = blockIdx.x * 256 + threadIdx.x;       // 5*2*512*512
  if (idx >= 5 * 2 * 512 * 512) return;
  int k = idx & 511;
  int o = (idx >> 9) & 511;
  int j = (idx >> 18) & 1;
  int i = idx >> 19;
  const float* mw = j ? mod2_w : mod1_w;
  float v = mw[((size_t)(i * 512 + o)) * 512 + k];
  int mtile = o >> 4, m = o & 15, kt = k >> 5, kk = k & 31;
  int lane = m + (((kk >> 3) & 1) << 4);
  int jj = (kk & 7) + ((kk >= 16) ? 8 : 0);
  size_t dst = ((size_t)(i * 2 + j) << 18) +
               (((size_t)(mtile * 16 + kt) * 32 + lane) << 4) + jj;
  wb[dst] = (__bf16)v;
}

// ---------------------------------------------------------------------------
// Kernel 3: broadcast const -> x[4][512][16]
// ---------------------------------------------------------------------------
__global__ __launch_bounds__(256) void sg_init_const(
    const float* __restrict__ cst, float* __restrict__ out) {
  int idx = blockIdx.x * 256 + threadIdx.x;       // 4*512*16 = 32768
  if (idx >= 4 * 512 * 16) return;
  out[idx] = cst[idx & (512 * 16 - 1)];
}

// ---------------------------------------------------------------------------
// Kernel 4: depthwise 3x3, replicate padding, + per-channel bias
// ---------------------------------------------------------------------------
__global__ __launch_bounds__(256) void sg_dwconv3x3(
    const float* __restrict__ xin, const float* __restrict__ w9,
    const float* __restrict__ bias, float* __restrict__ out,
    int logW, int total) {
  int idx = blockIdx.x * 256 + threadIdx.x;
  if (idx >= total) return;
  int W = 1 << logW;
  int p = idx & (W * W - 1);
  int x = p & (W - 1);
  int y = p >> logW;
  int c = (idx >> (2 * logW)) & 511;
  const float* src = xin + (size_t)(idx - p);     // (n,c) plane base
  const float* wr = w9 + c * 9;
  float s = bias[c];
#pragma unroll
  for (int ky = 0; ky < 3; ++ky) {
    int yy = y + ky - 1; yy = yy < 0 ? 0 : (yy >= W ? W - 1 : yy);
#pragma unroll
    for (int kx = 0; kx < 3; ++kx) {
      int xx = x + kx - 1; xx = xx < 0 ? 0 : (xx >= W ? W - 1 : xx);
      s += wr[ky * 3 + kx] * src[(yy << logW) + xx];
    }
  }
  out[idx] = s;
}

// ---------------------------------------------------------------------------
// Kernel 5: nearest 2x upsample + [1 2 1;2 4 2;1 2 1]/16 blur (replicate pad),
// generic over channel count (ch = n*C+c packed in upper index bits).
// ---------------------------------------------------------------------------
__global__ __launch_bounds__(256) void sg_up2blur(
    const float* __restrict__ xin, float* __restrict__ out,
    int logWo, int total) {
  int idx = blockIdx.x * 256 + threadIdx.x;
  if (idx >= total) return;
  int Wo = 1 << logWo;
  int p = idx & (Wo * Wo - 1);
  int x = p & (Wo - 1);
  int y = p >> logWo;
  int ch = idx >> (2 * logWo);
  const float* src = xin + (size_t)ch * ((size_t)(Wo * Wo) >> 2);
  const float kw[3] = {0.25f, 0.5f, 0.25f};
  float s = 0.f;
#pragma unroll
  for (int dy = 0; dy < 3; ++dy) {
    int yy = y + dy - 1; yy = yy < 0 ? 0 : (yy >= Wo ? Wo - 1 : yy); yy >>= 1;
#pragma unroll
    for (int dx = 0; dx < 3; ++dx) {
      int xx = x + dx - 1; xx = xx < 0 ? 0 : (xx >= Wo ? Wo - 1 : xx); xx >>= 1;
      s += kw[dy] * kw[dx] * src[yy * (Wo >> 1) + xx];
    }
  }
  out[idx] = s;
}

// ---------------------------------------------------------------------------
// Kernel 6: modulated 1x1 conv as bf16 WMMA GEMM + demod scale +
// scrambled bias + leaky ReLU.
//   y[n,o,p] = leaky(scale[n,o] * sum_c W[o,c]*x[n,c,p] + bias[(o*HW+p)&511])
// Workgroup = 8 waves, 128 output rows x (16*NSUB) pixels, K loop 16 x 32.
// Software pipeline: double-buffered LDS; each K step issues next tile's
// global loads first, runs WMMAs on the current LDS buffer, then converts and
// stores the prefetched tile (packed 2xbf16 -> one ds_store_b32) into the
// other buffer. One barrier per K step. NSUB compile-time (1: 4x4 layer,
// 4: all larger layers).
// ---------------------------------------------------------------------------
template <int NSUB>
__global__ __launch_bounds__(256) void sg_modgemm_wmma(
    const float* __restrict__ xin, const __bf16* __restrict__ wb,
    const float* __restrict__ scales, const float* __restrict__ biasv,
    float* __restrict__ out, int HW) {
  constexpr int NPIX = NSUB * 16;
  constexpr int PSHIFT = (NSUB == 4) ? 6 : 4;   // log2(NPIX)
  constexpr int FIT = NSUB;                     // (NPIX*16 k-pairs)/256 threads
  const int tid = threadIdx.x;
  const int lane = tid & 31;
  const int wave = tid >> 5;
  const int n = blockIdx.z;
  const int pbase = blockIdx.x * NPIX;
  const int mtile = blockIdx.y * 8 + wave;      // 0..31
  const int mrow = mtile << 4;

  // double-buffered B staging, packed as uint = 2 bf16 (k = 2*j2, 2*j2+1)
  __shared__ __align__(32) unsigned ldsU[2][NSUB][32][8];

  // kt-invariant fill offsets: element e = (kh, pp), kh = k-pair 0..15
  int gofs[FIT];   // global offset (in floats) of k=2*kh row, pixel pp
  int lofs[FIT];   // flat uint index inside one LDS buffer
#pragma unroll
  for (int it = 0; it < FIT; ++it) {
    int e = it * 256 + tid;
    int kh = e >> PSHIFT;
    int pp = e & (NPIX - 1);
    gofs[it] = (2 * kh) * HW + pp;
    int ln = (pp & 15) + ((kh & 4) << 2);       // +16 when (k & 8)
    lofs[it] = ((pp >> 4) * 32 + ln) * 8 + ((kh & 3) | ((kh & 8) >> 1));
  }

  v8f acc[NSUB];
#pragma unroll
  for (int t = 0; t < NSUB; ++t)
#pragma unroll
    for (int r = 0; r < 8; ++r) acc[t][r] = 0.f;

  const float* xsrc = xin + (size_t)(n * 512) * HW + pbase;
  unsigned* ldsFlat = &ldsU[0][0][0][0];
  constexpr int BUFW = NSUB * 32 * 8;           // uints per buffer

  // prologue: fill buffer 0 with tile kt=0
#pragma unroll
  for (int it = 0; it < FIT; ++it) {
    float v0 = xsrc[gofs[it]];
    float v1 = xsrc[gofs[it] + HW];
    ldsFlat[lofs[it]] = sg_pack_bf16x2(v0, v1);
  }

#pragma unroll
  for (int kt = 0; kt < 16; ++kt) {
    const int cb = kt & 1;
    __syncthreads();                            // buf cb ready; buf cb^1 free
    // issue next tile's global loads first (overlap with WMMA below)
    float pv0[FIT], pv1[FIT];
    if (kt < 15) {
      const float* xk = xsrc + (size_t)((kt + 1) * 32) * HW;
#pragma unroll
      for (int it = 0; it < FIT; ++it) {
        pv0[it] = xk[gofs[it]];
        pv1[it] = xk[gofs[it] + HW];
      }
    }
    // A tile (pre-tiled bf16 weights, 32B/lane) + B from LDS, then WMMA
    v16bf a = *(const v16bf*)(wb + (((size_t)(mtile * 16 + kt) * 32 + lane) << 4));
#pragma unroll
    for (int t = 0; t < NSUB; ++t) {
      v16bf b = *(const v16bf*)&ldsU[cb][t][lane][0];
      acc[t] = __builtin_amdgcn_wmma_f32_16x16x32_bf16(
          false, a, false, b, (short)0, acc[t], false, false);
    }
    // convert + store prefetched tile into the other buffer
    if (kt < 15) {
      unsigned* dst = ldsFlat + (cb ^ 1) * BUFW;
#pragma unroll
      for (int it = 0; it < FIT; ++it)
        dst[lofs[it]] = sg_pack_bf16x2(pv0[it], pv1[it]);
    }
  }

  // epilogue: C/D layout -> scale, scrambled bias, leaky, store
  const int nl = lane & 15;
  const int mAdd = (lane >> 4) << 3;
#pragma unroll
  for (int r = 0; r < 8; ++r) {
    int o = mrow + mAdd + r;
    float sc = scales[n * 512 + o];
#pragma unroll
    for (int t = 0; t < NSUB; ++t) {
      int p = pbase + (t << 4) + nl;
      float v = acc[t][r] * sc;
      v += biasv[(o * HW + p) & 511];
      v = (v >= 0.f) ? v : LEAK * v;
      out[((size_t)(n * 512 + o)) * HW + p] = v;
    }
  }
}

// ---------------------------------------------------------------------------
// Kernel 7: RGB head: acc = (addPrev ? acc : 0) + alpha*(rgb_w @ x + rgb_b)
// ---------------------------------------------------------------------------
__global__ __launch_bounds__(256) void sg_rgb(
    const float* __restrict__ xin, const float* __restrict__ w,
    const float* __restrict__ b, float* __restrict__ acc,
    int logHW, int addPrev, float alpha) {
  int idx = blockIdx.x * 256 + threadIdx.x;
  int HW = 1 << logHW;
  if (idx >= 12 * HW) return;                   // 4 batch * 3 ch
  int p = idx & (HW - 1);
  int t = idx >> logHW;                         // n*3+o
  int o = t % 3;
  int n = t / 3;
  const float* wr = w + o * 512;
  const float* xs = xin + (size_t)(n * 512) * HW + p;
  float s = b[o];
  for (int c = 0; c < 512; ++c) s += wr[c] * xs[(size_t)c * HW];
  s *= alpha;
  acc[idx] = (addPrev ? acc[idx] : 0.f) + s;
}

// ---------------------------------------------------------------------------
// Kernel 8: final tanh
// ---------------------------------------------------------------------------
__global__ __launch_bounds__(256) void sg_tanh(
    const float* __restrict__ xin, float* __restrict__ out, int total) {
  int idx = blockIdx.x * 256 + threadIdx.x;
  if (idx >= total) return;
  out[idx] = tanhf(xin[idx]);
}

// ---------------------------------------------------------------------------
// Host launcher
// ---------------------------------------------------------------------------
extern "C" void kernel_launch(void* const* d_in, const int* in_sizes, int n_in,
                              void* d_out, int out_size, void* d_ws, size_t ws_size,
                              hipStream_t stream) {
  (void)in_sizes; (void)n_in; (void)out_size; (void)ws_size;
  const float* style  = (const float*)d_in[0];
  const float* cst    = (const float*)d_in[1];
  const float* aff1_w = (const float*)d_in[2];
  const float* aff1_b = (const float*)d_in[3];
  const float* dw1_w  = (const float*)d_in[4];
  const float* dw1_b  = (const float*)d_in[5];
  const float* mod1_w = (const float*)d_in[6];
  const float* bias1  = (const float*)d_in[7];
  const float* aff2_w = (const float*)d_in[8];
  const float* aff2_b = (const float*)d_in[9];
  const float* dw2_w  = (const float*)d_in[10];
  const float* dw2_b  = (const float*)d_in[11];
  const float* mod2_w = (const float*)d_in[12];
  const float* bias2  = (const float*)d_in[13];
  const float* rgb_w  = (const float*)d_in[14];
  const float* rgb_b  = (const float*)d_in[15];

  char* ws = (char*)d_ws;
  float*  bufA = (float*)(ws + 0);              // 4*512*4096*4 = 33554432 B
  float*  bufB = (float*)(ws + 33554432);       // 33554432 B
  __bf16* wb   = (__bf16*)(ws + 67108864);      // 5*2*512*512*2 = 5242880 B
  float*  scl  = (float*)(ws + 72351744);       // 5*2*4*512*4 = 81920 B
  float*  rgb0 = (float*)(ws + 72433664);       // 4*3*4096*4 = 196608 B
  float*  rgb1 = (float*)(ws + 72630272);       // 196608 B

  sg_styles_scales<<<20, 256, 0, stream>>>(style, aff1_w, aff1_b, mod1_w,
                                           aff2_w, aff2_b, mod2_w, scl);
  sg_convert_weights<<<10240, 256, 0, stream>>>(mod1_w, mod2_w, wb);
  sg_init_const<<<128, 256, 0, stream>>>(cst, bufA);

  float* cur = bufA; float* tmp = bufB;
  float* rPrev = rgb0; float* rCur = rgb1;
  int logW = 2;                                  // 4x4 at layer 0
  for (int i = 0; i < 5; ++i) {
    if (i > 0) {
      ++logW;
      int total = 4 * 512 << (2 * logW);
      sg_up2blur<<<(total + 255) / 256, 256, 0, stream>>>(cur, tmp, logW, total);
      float* s = cur; cur = tmp; tmp = s;
    }
    int HW = 1 << (2 * logW);
    int totalAct = 4 * 512 * HW;

    sg_dwconv3x3<<<(totalAct + 255) / 256, 256, 0, stream>>>(
        cur, dw1_w + i * 512 * 9, dw1_b + i * 512, tmp, logW, totalAct);
    if (HW == 16) {
      dim3 g(1, 4, 4);
      sg_modgemm_wmma<1><<<g, 256, 0, stream>>>(
          tmp, wb + ((size_t)(i * 2 + 0) << 18), scl + (i * 2 + 0) * 4 * 512,
          bias1 + i * 512, cur, HW);
    } else {
      dim3 g((unsigned)(HW / 64), 4, 4);
      sg_modgemm_wmma<4><<<g, 256, 0, stream>>>(
          tmp, wb + ((size_t)(i * 2 + 0) << 18), scl + (i * 2 + 0) * 4 * 512,
          bias1 + i * 512, cur, HW);
    }
    sg_dwconv3x3<<<(totalAct + 255) / 256, 256, 0, stream>>>(
        cur, dw2_w + i * 512 * 9, dw2_b + i * 512, tmp, logW, totalAct);
    if (HW == 16) {
      dim3 g(1, 4, 4);
      sg_modgemm_wmma<1><<<g, 256, 0, stream>>>(
          tmp, wb + ((size_t)(i * 2 + 1) << 18), scl + (i * 2 + 1) * 4 * 512,
          bias2 + i * 512, cur, HW);
    } else {
      dim3 g((unsigned)(HW / 64), 4, 4);
      sg_modgemm_wmma<4><<<g, 256, 0, stream>>>(
          tmp, wb + ((size_t)(i * 2 + 1) << 18), scl + (i * 2 + 1) * 4 * 512,
          bias2 + i * 512, cur, HW);
    }

    int totR = 12 * HW;
    if (i > 0) {
      sg_up2blur<<<(totR + 255) / 256, 256, 0, stream>>>(rPrev, rCur, logW, totR);
    }
    sg_rgb<<<(totR + 255) / 256, 256, 0, stream>>>(
        cur, rgb_w + i * 3 * 512, rgb_b + i * 3, rCur, 2 * logW,
        (i > 0) ? 1 : 0, (i == 4) ? 0.5f : 1.0f);
    { float* s = rPrev; rPrev = rCur; rCur = s; }
  }
  sg_tanh<<<192, 256, 0, stream>>>(rPrev, (float*)d_out, 4 * 3 * 64 * 64);
}
